// DotAttention_63153199120804
// MI455X (gfx1250) — compile-verified, matrix operand order checked by reading, not run on hardware
//
#include <hip/hip_runtime.h>
#include <hip/hip_bf16.h>
#include <math.h>

typedef __bf16 bf16;
typedef __attribute__((ext_vector_type(16))) __bf16 v16bf;
typedef __attribute__((ext_vector_type(8)))  __bf16 v8bf;
typedef __attribute__((ext_vector_type(4)))  __bf16 v4bf;
typedef __attribute__((ext_vector_type(8)))  float  v8f;
typedef __attribute__((ext_vector_type(4)))  float  v4f;

#define WMMA_BF16(a, b, c) \
  __builtin_amdgcn_wmma_f32_16x16x32_bf16(false, (a), false, (b), (short)0, (c), false, false)

// Load one 16x32 (16-bit) WMMA A/B fragment row-region.
// `row` points at the 32 contiguous K-elements for this lane's matrix row/col.
// Element e of the fragment holds K = (e<8?0:16) + 8*half + (e&7)  (ISA 7.12.2),
// i.e. two contiguous 8-element (16B) chunks at K-offsets 8*half and 16+8*half.
__device__ __forceinline__ v16bf load_frag(const bf16* row, int half) {
  v8bf lo = *(const v8bf*)(row + 8 * half);
  v8bf hi = *(const v8bf*)(row + 16 + 8 * half);
  v16bf r;
#pragma unroll
  for (int i = 0; i < 8; ++i) { r[i] = lo[i]; r[i + 8] = hi[i]; }
  return r;
}

// ---------------------------------------------------------------------------
// Generic WMMA GEMM:  C[M,N] = (A[M,K] @ W[K,N] + bias) * out_scale
// A is fp32 (converted to bf16 on the fly) or bf16; C is bf16 or fp32.
// Block: 256 threads = 8 waves; 128x128 tile; BK=64 (16 WMMAs per wave between
// barriers); wave grid 4x2. Next K-slab prefetched via global_prefetch.
// ---------------------------------------------------------------------------
template <bool A_F32, bool OUT_BF16>
__global__ __launch_bounds__(256, 2) void gemm_bias_kernel(
    const void* __restrict__ Ap, const float* __restrict__ W,
    const float* __restrict__ bias, void* __restrict__ Cp,
    int M, int N, int K, float out_scale) {
  constexpr int LS = 72;  // LDS row stride in bf16 (64 + 8 pad, keeps 16B align)
  __shared__ __align__(16) bf16 ldsA[128 * LS];   // A tile, row-major [m][k]
  __shared__ __align__(16) bf16 ldsBT[128 * LS];  // W tile, transposed [n][k]

  const int tid  = threadIdx.x;
  const int lane = tid & 31;
  const int wv   = tid >> 5;
  const int half = lane >> 4;
  const int ln   = lane & 15;
  const int wr   = wv >> 1;  // 0..3 -> 32-row slab
  const int wc   = wv & 1;   // 0..1 -> 64-col slab
  const int m0   = blockIdx.y * 128;
  const int n0   = blockIdx.x * 128;

  v8f acc[2][4];
#pragma unroll
  for (int i = 0; i < 2; ++i)
#pragma unroll
    for (int j = 0; j < 4; ++j) acc[i][j] = (v8f)(0.0f);

  // staging coordinates
  const int arow = tid >> 4;         // 0..15  (A: 8 passes of 16 rows)
  const int acol = (tid & 15) * 4;   // 0..60
  const int bk   = tid >> 5;         // 0..7   (W: 8 passes of 8 k-rows)
  const int bn   = lane * 4;         // 0..124

  for (int kk = 0; kk < K; kk += 64) {
    // --- stage A tile (128x64), converting f32->bf16 if needed ---
#pragma unroll
    for (int p = 0; p < 8; ++p) {
      int r = arow + p * 16;
      if (A_F32) {
        const float* A = (const float*)Ap;
        v4f x = *(const v4f*)(A + (size_t)(m0 + r) * K + kk + acol);
        bf16* d = &ldsA[r * LS + acol];
        d[0] = (bf16)x[0]; d[1] = (bf16)x[1];
        d[2] = (bf16)x[2]; d[3] = (bf16)x[3];
      } else {
        const bf16* A = (const bf16*)Ap;
        *(v4bf*)&ldsA[r * LS + acol] =
            *(const v4bf*)(A + (size_t)(m0 + r) * K + kk + acol);
      }
    }
    // --- stage W tile (64x128) transposed into ldsBT[n][k] ---
#pragma unroll
    for (int p = 0; p < 8; ++p) {
      int kr = bk + p * 8;
      v4f x = *(const v4f*)(W + (size_t)(kk + kr) * N + n0 + bn);
#pragma unroll
      for (int i = 0; i < 4; ++i) ldsBT[(bn + i) * LS + kr] = (bf16)x[i];
    }
    // --- prefetch next K-slab while this one computes ---
    if (kk + 64 < K) {
      if (A_F32)
        __builtin_prefetch((const float*)Ap + (size_t)(m0 + arow) * K + kk + 64 + acol, 0, 1);
      else
        __builtin_prefetch((const bf16*)Ap + (size_t)(m0 + arow) * K + kk + 64 + acol, 0, 1);
      __builtin_prefetch(W + (size_t)(kk + 64 + bk) * N + n0 + bn, 0, 1);
    }
    __syncthreads();

#pragma unroll
    for (int ks = 0; ks < 64; ks += 32) {
      v16bf bfr[4];
#pragma unroll
      for (int j = 0; j < 4; ++j)
        bfr[j] = load_frag(&ldsBT[(wc * 64 + j * 16 + ln) * LS + ks], half);
#pragma unroll
      for (int i = 0; i < 2; ++i) {
        v16bf afr = load_frag(&ldsA[(wr * 32 + i * 16 + ln) * LS + ks], half);
#pragma unroll
        for (int j = 0; j < 4; ++j)
          acc[i][j] = WMMA_BF16(afr, bfr[j], acc[i][j]);
      }
    }
    __syncthreads();
  }

  // C layout (ISA 7.12.2): lane -> col n = ln (+half selects M+8), VGPR r -> row
#pragma unroll
  for (int i = 0; i < 2; ++i) {
#pragma unroll
    for (int j = 0; j < 4; ++j) {
      int n = n0 + wc * 64 + j * 16 + ln;
      float bv = bias[n];
#pragma unroll
      for (int r = 0; r < 8; ++r) {
        int m = m0 + wr * 32 + i * 16 + r + 8 * half;
        float v = (acc[i][j][r] + bv) * out_scale;
        if (OUT_BF16) ((bf16*)Cp)[(size_t)m * N + n] = (bf16)v;
        else          ((float*)Cp)[(size_t)m * N + n] = v;
      }
    }
  }
}

// ---------------------------------------------------------------------------
// Flash attention: one block = (batch b, head h, 128 queries); 8 waves, each
// owns 16 queries. KV processed in tiles of 32 keys staged in LDS.
// S^T = K_tile x Q^T (keys on VGPRs, queries on lanes): softmax reduces over
// in-lane VGPRs + one xor-16 shuffle; the two 16-key S^T C-tiles repack
// in-lane into the K=32 B-fragment of P for O^T = V^T x P^T.
// KV loop stops at ceil(vlen/32): tiles fully beyond vlen contribute exactly
// zero in fp32 (exp(qk-1e30-m) underflows). vlen==0 -> full masked loop,
// reproducing the reference's uniform-softmax behavior.
// ---------------------------------------------------------------------------
__global__ __launch_bounds__(256, 2) void flash_attn_kernel(
    const bf16* __restrict__ Q, const bf16* __restrict__ Kb,
    const bf16* __restrict__ Vb, const int* __restrict__ vlens,
    bf16* __restrict__ Ctx, int Tq, int Tv) {
  constexpr int D  = 1024;  // model dim
  constexpr int DH = 64;    // head dim
  constexpr int KS = 72;    // ldsK row stride (bf16): 64 + 8 pad, 16B aligned
  constexpr int VS = 40;    // ldsVT row stride:       32 + 8 pad, 16B aligned
  __shared__ __align__(16) bf16 ldsK[32 * KS];   // [key][d]
  __shared__ __align__(16) bf16 ldsVT[64 * VS];  // [d][key] (transposed)

  const int tid  = threadIdx.x;
  const int lane = tid & 31;
  const int wv   = tid >> 5;
  const int half = lane >> 4;
  const int ln   = lane & 15;
  const int b    = blockIdx.z;
  const int h    = blockIdx.y;
  const int q0   = blockIdx.x * 128 + wv * 16;
  const int vlen = vlens[b];

  // Q fragments (B operand for S^T; lane = query column): stay in registers.
  const size_t qrow = ((size_t)b * Tq + q0 + ln) * D + (size_t)h * DH;
  v16bf qf0 = load_frag(Q + qrow, half);       // d 0..31
  v16bf qf1 = load_frag(Q + qrow + 32, half);  // d 32..63

  v8f o[4];
#pragma unroll
  for (int t = 0; t < 4; ++t) o[t] = (v8f)(0.0f);
  float mrun = -1e38f, lrun = 0.0f;

  int kv_end = (vlen == 0) ? Tv : ((vlen + 31) & ~31);
  if (kv_end > Tv) kv_end = Tv;

  // per-thread staging coords
  const int sk = tid >> 3;        // 0..31 (key)
  const int sd = (tid & 7) * 8;   // 0..56 (d chunk)

  for (int kv0 = 0; kv0 < kv_end; kv0 += 32) {
    // --- cooperative stage: K tile direct, V tile transposed ---
    {
      const size_t base = ((size_t)b * Tv + kv0 + sk) * D + (size_t)h * DH + sd;
      *(v8bf*)&ldsK[sk * KS + sd] = *(const v8bf*)(Kb + base);
      v8bf vv = *(const v8bf*)(Vb + base);
#pragma unroll
      for (int i = 0; i < 8; ++i) ldsVT[(sd + i) * VS + sk] = vv[i];
      if (kv0 + 32 < kv_end) {  // prefetch next KV tile
        __builtin_prefetch(Kb + base + (size_t)32 * D, 0, 1);
        __builtin_prefetch(Vb + base + (size_t)32 * D, 0, 1);
      }
    }
    __syncthreads();

    // --- S^T tiles: keys 0..15 (s0) and 16..31 (s1), K-dim = d (2 steps) ---
    v8f z = (v8f)(0.0f);
    v16bf ka0 = load_frag(&ldsK[ln * KS], half);
    v16bf ka1 = load_frag(&ldsK[ln * KS + 32], half);
    v8f s0 = WMMA_BF16(ka0, qf0, z);
    s0 = WMMA_BF16(ka1, qf1, s0);
    v16bf kb0 = load_frag(&ldsK[(16 + ln) * KS], half);
    v16bf kb1 = load_frag(&ldsK[(16 + ln) * KS + 32], half);
    v8f s1 = WMMA_BF16(kb0, qf0, z);
    s1 = WMMA_BF16(kb1, qf1, s1);

    // --- mask only on the boundary (or fully-masked, vlen==0) tile ---
    if (kv0 + 32 > vlen) {
#pragma unroll
      for (int r = 0; r < 8; ++r) {
        int key0 = kv0 + r + 8 * half;  // s0 VGPR r holds key r + 8*half
        if (key0 >= vlen)      s0[r] -= 1e30f;
        if (key0 + 16 >= vlen) s1[r] -= 1e30f;
      }
    }

    // --- row max (over in-lane VGPRs, then partner half-wave) ---
    float mx = -1e38f;
#pragma unroll
    for (int r = 0; r < 8; ++r) mx = fmaxf(mx, fmaxf(s0[r], s1[r]));
    mx = fmaxf(mx, __shfl_xor(mx, 16, 32));
    float mnew  = fmaxf(mrun, mx);
    float scale = __expf(mrun - mnew);
    mrun = mnew;

    // --- P = exp(S - mnew); repack in-lane into K=32 B-fragment ---
    v16bf pf;
    float ls = 0.0f;
#pragma unroll
    for (int r = 0; r < 8; ++r) {
      float e0 = __expf(s0[r] - mnew);
      float e1 = __expf(s1[r] - mnew);
      ls += e0 + e1;
      pf[r]     = (bf16)e0;  // B elem e<8  -> key (e&7)+8*half    = s0[r]
      pf[r + 8] = (bf16)e1;  // B elem e>=8 -> key 16+(e&7)+8*half = s1[r]
    }
    lrun = lrun * scale + ls;

    // --- O^T[d,q] accumulate: A = V^T rows (16 d per tile, K = 32 keys) ---
#pragma unroll
    for (int t = 0; t < 4; ++t) {
#pragma unroll
      for (int r = 0; r < 8; ++r) o[t][r] *= scale;
      v16bf va = load_frag(&ldsVT[(t * 16 + ln) * VS], half);
      o[t] = WMMA_BF16(va, pf, o[t]);
    }
    __syncthreads();
  }

  // Each lane's lrun covers 16 of every 32 keys; partner lane has the rest.
  float ltot = lrun + __shfl_xor(lrun, 16, 32);
  float inv  = 1.0f / ltot;
#pragma unroll
  for (int t = 0; t < 4; ++t) {
    v8bf ov;
#pragma unroll
    for (int r = 0; r < 8; ++r) ov[r] = (bf16)(o[t][r] * inv);
    // O^T C layout: lane -> query ln; VGPR r -> d = 16*t + r + 8*half (contig 8)
    bf16* dst = Ctx + ((size_t)b * Tq + q0 + ln) * D + (size_t)h * DH +
                t * 16 + 8 * half;
    *(v8bf*)dst = ov;
  }
}

// ---------------------------------------------------------------------------
extern "C" void kernel_launch(void* const* d_in, const int* in_sizes, int n_in,
                              void* d_out, int out_size, void* d_ws,
                              size_t ws_size, hipStream_t stream) {
  const float* query = (const float*)d_in[0];
  const float* value = (const float*)d_in[1];
  const int*   vlens = (const int*)d_in[2];
  const float* Wq = (const float*)d_in[3];
  const float* bq = (const float*)d_in[4];
  const float* Wk = (const float*)d_in[5];
  const float* bk = (const float*)d_in[6];
  const float* Wv = (const float*)d_in[7];
  const float* bv = (const float*)d_in[8];
  const float* Wf = (const float*)d_in[9];
  const float* bfb = (const float*)d_in[10];

  const int B = 4, Tq = 2048, Tv = 2048, Dm = 1024;
  const int M = B * Tq;  // 8192

  size_t seg = (size_t)M * Dm * sizeof(bf16);  // 16 MB per bf16 tensor
  bf16* Qbf = (bf16*)d_ws;
  bf16* Kbf = (bf16*)((char*)d_ws + seg);
  bf16* Vbf = (bf16*)((char*)d_ws + 2 * seg);
  bf16* Ctx = (bf16*)((char*)d_ws + 3 * seg);

  dim3 blk(256);
  dim3 gg(Dm / 128, M / 128);  // (8, 64)

  // Projections (bf16 out). 1/sqrt(d_head)=0.125 folded into Q epilogue.
  hipLaunchKernelGGL((gemm_bias_kernel<true, true>), gg, blk, 0, stream,
                     (const void*)query, Wq, bq, (void*)Qbf, M, Dm, Dm, 0.125f);
  hipLaunchKernelGGL((gemm_bias_kernel<true, true>), gg, blk, 0, stream,
                     (const void*)value, Wk, bk, (void*)Kbf, M, Dm, Dm, 1.0f);
  hipLaunchKernelGGL((gemm_bias_kernel<true, true>), gg, blk, 0, stream,
                     (const void*)value, Wv, bv, (void*)Vbf, M, Dm, Dm, 1.0f);

  // Flash attention: grid (q-tiles, heads, batch)
  dim3 ga(Tq / 128, 16, B);
  hipLaunchKernelGGL(flash_attn_kernel, ga, blk, 0, stream, Qbf, Kbf, Vbf,
                     vlens, Ctx, Tq, Tv);

  // Output projection: ctx(bf16) @ Wf + bf -> fp32
  hipLaunchKernelGGL((gemm_bias_kernel<false, false>), gg, blk, 0, stream,
                     (const void*)Ctx, Wf, bfb, d_out, M, Dm, Dm, 1.0f);
}
